// LSTM_77481210020575
// MI455X (gfx1250) — compile-verified, matrix operand order checked by reading, not run on hardware
//
#include <hip/hip_runtime.h>
#include <hip/hip_bf16.h>

// ---------------- problem dims ----------------
#define T_STEPS 1024
#define B_DIM   64
#define I_DIM   256
#define H_DIM   512
#define G4H     2048            // 4*H
#define KTOT    (I_DIM + H_DIM) // 768

// ---------------- partitioning ----------------
#define NWG         32          // workgroups; each owns H/NWG h-columns
#define COLS_PER_WG 16          // one 16-wide N tile per gate
#define NTHREADS    128         // 4 waves; wave w -> M tile w (B = 4*16)
#define LDS_ELEMS   (4 * COLS_PER_WG * KTOT)   // 64 cols * 768 K (bf16) = 96KB

typedef __attribute__((ext_vector_type(16))) __bf16       v16bf;
typedef __attribute__((ext_vector_type(8)))  float        v8f;
typedef __attribute__((ext_vector_type(4)))  unsigned int u32x4;
typedef __attribute__((ext_vector_type(4)))  float        f32x4;

union ABFrag {
    v16bf  v;
    u32x4  q[2];
    __bf16 h[16];
};

__device__ __forceinline__ float sigmoid_f(float x) {
    return 1.0f / (1.0f + __expf(-x));
}
__device__ __forceinline__ float tanh_f(float x) {
    float e = __expf(-2.0f * x);
    return (1.0f - e) / (1.0f + e);
}

// Prologue: zero the grid-barrier counter, seed h double-buffer with h0 (bf16).
__global__ void lstm_init(const float* __restrict__ h0,
                          unsigned int* counter,
                          __hip_bfloat16* hbuf) {
    int idx = blockIdx.x * blockDim.x + threadIdx.x;
    if (idx == 0) *counter = 0u;
    for (int i = idx; i < B_DIM * H_DIM; i += gridDim.x * blockDim.x) {
        hbuf[i] = __float2bfloat16(h0[i]);                 // buffer 0 = h0
        hbuf[B_DIM * H_DIM + i] = __float2bfloat16(0.0f);  // buffer 1
    }
}

__global__ __launch_bounds__(NTHREADS)
void lstm_scan(const float* __restrict__ x,    // [T,B,I]
               const float* __restrict__ Wi,   // [I,4H]
               const float* __restrict__ Wh,   // [H,4H]
               const float* __restrict__ bi,   // [4H]
               const float* __restrict__ bh,   // [4H]
               const float* __restrict__ c0,   // [B,H]
               float* __restrict__ hs,         // [T,B,H] output
               unsigned int* counter,
               __hip_bfloat16* hbuf)           // [2][B,H] bf16 double buffer
{
    __shared__ __hip_bfloat16 ldsW[LDS_ELEMS]; // [64 local cols][768 K], K contiguous

    const int tid  = threadIdx.x;
    const int wg   = blockIdx.x;
    const int lane = tid & 31;
    const int wave = tid >> 5;

    // ---- one-time: stage this WG's bf16 weight slice (Wi;Wh stacked on K) ----
    for (int e = tid; e < LDS_ELEMS; e += NTHREADS) {
        int k  = e >> 6;             // 0..767
        int cl = e & 63;             // local col: gate*16 + n
        int g  = cl >> 4;
        int n  = cl & 15;
        int gcol = (g << 9) + wg * COLS_PER_WG + n;     // column in [0,2048)
        float w = (k < I_DIM) ? Wi[(size_t)k * G4H + gcol]
                              : Wh[(size_t)(k - I_DIM) * G4H + gcol];
        ldsW[cl * KTOT + k] = __float2bfloat16(w);
    }
    __syncthreads();

    // ---- per-lane geometry (ISA C/D + A/B 16-bit VGPR layouts) ----
    const int myN     = lane & 15;                        // N within the 16-wide tile
    const int colH    = wg * COLS_PER_WG + myN;           // h/c column (0..511)
    const int loK     = (lane < 16) ? 0 : 8;              // A/B K sub-offset per half-wave
    const int rowHalf = (lane >> 4) << 3;                 // C/D: lanes 16..31 -> M+8
    const int rowBase = wave * 16 + rowHalf;              // C/D rows rowBase..rowBase+7
    const int arow    = wave * 16 + (lane & 15);          // A-matrix row for loads

    // bias per gate (same for all 8 accumulator rows of this lane)
    float biasg[4];
#pragma unroll
    for (int g = 0; g < 4; ++g)
        biasg[g] = bi[(g << 9) + colH] + bh[(g << 9) + colH];

    // cell state in C/D accumulator layout
    v8f c;
#pragma unroll
    for (int r = 0; r < 8; ++r)
        c[r] = c0[(size_t)(rowBase + r) * H_DIM + colH];

    for (int t = 0; t < T_STEPS; ++t) {
        const __hip_bfloat16* hcur = hbuf + (size_t)(t & 1) * (B_DIM * H_DIM);

        v8f acc[4];
#pragma unroll
        for (int g = 0; g < 4; ++g)
#pragma unroll
            for (int r = 0; r < 8; ++r) acc[g][r] = biasg[g];

        // ---- K over x_t (convert f32 -> bf16 on the fly) ----
        for (int kc = 0; kc < I_DIM; kc += 32) {
            ABFrag a;
            const float* xs = x + ((size_t)t * B_DIM + arow) * I_DIM + kc + loK;
            f32x4 x0 = *(const f32x4*)(xs);
            f32x4 x1 = *(const f32x4*)(xs + 4);
            f32x4 x2 = *(const f32x4*)(xs + 16);
            f32x4 x3 = *(const f32x4*)(xs + 20);
#pragma unroll
            for (int i = 0; i < 4; ++i) {
                a.h[i]      = (__bf16)x0[i];
                a.h[4 + i]  = (__bf16)x1[i];
                a.h[8 + i]  = (__bf16)x2[i];
                a.h[12 + i] = (__bf16)x3[i];
            }
            // Load all 4 gate B-fragments first (one ds clause), then issue the
            // 4 WMMAs -> partial dscnt waits instead of full serialization.
            ABFrag b[4];
#pragma unroll
            for (int g = 0; g < 4; ++g) {
                const __hip_bfloat16* wp = &ldsW[((g << 4) + myN) * KTOT + kc + loK];
                b[g].q[0] = *(const u32x4*)(wp);
                b[g].q[1] = *(const u32x4*)(wp + 16);
            }
#pragma unroll
            for (int g = 0; g < 4; ++g)
                acc[g] = __builtin_amdgcn_wmma_f32_16x16x32_bf16(
                    false, a.v, false, b[g].v, (short)0, acc[g], false, false);
        }

        // ---- K over h (already bf16 in the shared double buffer) ----
        for (int kc = 0; kc < H_DIM; kc += 32) {
            ABFrag a;
            const __hip_bfloat16* hp = hcur + (size_t)arow * H_DIM + kc + loK;
            a.q[0] = *(const u32x4*)(hp);
            a.q[1] = *(const u32x4*)(hp + 16);

            ABFrag b[4];
#pragma unroll
            for (int g = 0; g < 4; ++g) {
                const __hip_bfloat16* wp =
                    &ldsW[((g << 4) + myN) * KTOT + (I_DIM + kc) + loK];
                b[g].q[0] = *(const u32x4*)(wp);
                b[g].q[1] = *(const u32x4*)(wp + 16);
            }
#pragma unroll
            for (int g = 0; g < 4; ++g)
                acc[g] = __builtin_amdgcn_wmma_f32_16x16x32_bf16(
                    false, a.v, false, b[g].v, (short)0, acc[g], false, false);
        }

        // ---- gate nonlinearities + state update (pure per-lane VALU) ----
        __hip_bfloat16* hnxt = hbuf + (size_t)((t + 1) & 1) * (B_DIM * H_DIM);
        float* outp = hs + (size_t)t * B_DIM * H_DIM;
#pragma unroll
        for (int r = 0; r < 8; ++r) {
            float ig = sigmoid_f(acc[0][r]);
            float fg = sigmoid_f(acc[1][r]);
            float gg = tanh_f(acc[2][r]);
            float og = sigmoid_f(acc[3][r]);
            float cn = fg * c[r] + ig * gg;
            c[r] = cn;
            float hn = og * tanh_f(cn);
            int row = rowBase + r;
            outp[(size_t)row * H_DIM + colH] = hn;
            hnxt[(size_t)row * H_DIM + colH] = __float2bfloat16(hn);
        }

        // ---- grid-wide barrier: make h_{t+1} visible to all 32 WGs ----
        __builtin_amdgcn_fence(__ATOMIC_RELEASE, "agent");
        __syncthreads();
        if (tid == 0) {
            __hip_atomic_fetch_add(counter, 1u, __ATOMIC_RELAXED,
                                   __HIP_MEMORY_SCOPE_AGENT);
            const unsigned target = (unsigned)NWG * (unsigned)(t + 1);
            while (__hip_atomic_load(counter, __ATOMIC_RELAXED,
                                     __HIP_MEMORY_SCOPE_AGENT) < target)
                __builtin_amdgcn_s_sleep(1);
        }
        __syncthreads();
        __builtin_amdgcn_fence(__ATOMIC_ACQUIRE, "agent");
    }
}

extern "C" void kernel_launch(void* const* d_in, const int* in_sizes, int n_in,
                              void* d_out, int out_size, void* d_ws, size_t ws_size,
                              hipStream_t stream) {
    const float* x  = (const float*)d_in[0];
    const float* Wi = (const float*)d_in[1];
    const float* Wh = (const float*)d_in[2];
    const float* bi = (const float*)d_in[3];
    const float* bh = (const float*)d_in[4];
    const float* h0 = (const float*)d_in[5];
    const float* c0 = (const float*)d_in[6];
    float* hs = (float*)d_out;

    unsigned int*   counter = (unsigned int*)d_ws;
    __hip_bfloat16* hbuf    = (__hip_bfloat16*)((char*)d_ws + 256);

    lstm_init<<<64, 256, 0, stream>>>(h0, counter, hbuf);
    lstm_scan<<<NWG, NTHREADS, 0, stream>>>(x, Wi, Wh, bi, bh, c0, hs,
                                            counter, hbuf);
}
